// EncoderDecoder_64905545777713
// MI455X (gfx1250) — matrix operand order repaired and checked by reading, not compiled, not run on hardware
//
#include <hip/hip_runtime.h>
#include <hip/hip_bf16.h>
#include <math.h>

// Problem constants (match reference)
#define U_  512
#define A_  512
#define E_  300
#define VT_ 30000
#define S_  64
#define T_  32
#define B_  32

#define KP_E   320     // pad(300, 32)
#define KP_U   512
#define KP_2U  1024
#define KP_3U  1536
#define KP_2UE 1344    // pad(1324, 32)
#define KP_3UE 1856    // pad(1836, 32)

typedef __bf16 bf16_t;
typedef bf16_t v16bf __attribute__((ext_vector_type(16)));
typedef bf16_t v8bf  __attribute__((ext_vector_type(8)));
typedef float  v8f   __attribute__((ext_vector_type(8)));

// ---------------------------------------------------------------- utilities
__global__ void k_fill_f32(float* p, long n, float v) {
  long i = (long)blockIdx.x * blockDim.x + threadIdx.x;
  if (i < n) p[i] = v;
}

// fp32 (N x ld_in, starting at col_off) -> bf16 (N x Kpad), zero-padded
__global__ void k_cvt_bf16(const float* __restrict__ in, bf16_t* __restrict__ out,
                           int N, int K, int Kpad, int ld_in, int col_off) {
  long i = (long)blockIdx.x * blockDim.x + threadIdx.x;
  long tot = (long)N * Kpad;
  if (i >= tot) return;
  int n = (int)(i / Kpad), k = (int)(i % Kpad);
  float v = (k < K) ? in[(long)n * ld_in + col_off + k] : 0.f;
  out[i] = (bf16_t)v;
}

// embedding gather -> bf16 rows (ntok x Kpad)
__global__ void k_gather_emb(const float* __restrict__ emb, const int* __restrict__ tok,
                             bf16_t* __restrict__ out, int ntok, int Ecols, int Kpad) {
  long i = (long)blockIdx.x * blockDim.x + threadIdx.x;
  long tot = (long)ntok * Kpad;
  if (i >= tot) return;
  int n = (int)(i / Kpad), k = (int)(i % Kpad);
  float v = (k < Ecols) ? emb[(long)tok[n] * Ecols + k] : 0.f;
  out[i] = (bf16_t)v;
}

// ---------------------------------------------------------------- WMMA GEMM
// C[M,N](f32,ldc) = A[M,Kpad](bf16,lda) @ W[N,Kpad](bf16,ldw)^T + bias; act=1 -> tanh
// Each wave owns BOTH M-tiles of a 32-row block (M multiple of 32) for one 16-col
// strip: one B-fragment load feeds two WMMAs -> weight (L2) traffic halved.
// 8 waves per block cover 128 columns.
__global__ __launch_bounds__(256)
void k_gemm_bf16(const bf16_t* __restrict__ A, int lda,
                 const bf16_t* __restrict__ W, int ldw,
                 float* __restrict__ C, int ldc,
                 const float* __restrict__ bias,
                 int M, int N, int Kpad, int act) {
  int lane = threadIdx.x & 31;
  int wave = threadIdx.x >> 5;                  // 8 N-tiles per block
  int m0 = blockIdx.y * 32;
  int n0 = blockIdx.x * 128 + wave * 16;
  if (m0 >= M || n0 >= N) return;               // wave-uniform: EXEC stays all-ones
  int half = lane >> 4;                         // 0: lanes 0-15, 1: lanes 16-31
  int l15  = lane & 15;
  const bf16_t* arow0 = A + (size_t)(m0 + l15) * lda;        // A rows m0..m0+15
  const bf16_t* arow1 = A + (size_t)(m0 + 16 + l15) * lda;   // A rows m0+16..m0+31
  const bf16_t* wrow  = W + (size_t)(n0 + l15) * ldw;        // B col: N = n0+l15
  v8f acc0 = {}, acc1 = {};
  for (int k0 = 0; k0 < Kpad; k0 += 32) {
    union { v16bf v; v8bf h[2]; } ua0, ua1, ub;
    // 16-bit B 32x16 layout: lanes0-15 hold K=0..7,16..23; lanes16-31 K=8..15,24..31
    ub.h[0]  = *(const v8bf*)(wrow + k0 + half * 8);
    ub.h[1]  = *(const v8bf*)(wrow + k0 + 16 + half * 8);
    // 16-bit A 16x32 layout: lanes0-15 hold K=0..7,16..23; lanes16-31 K=8..15,24..31
    ua0.h[0] = *(const v8bf*)(arow0 + k0 + half * 8);
    ua0.h[1] = *(const v8bf*)(arow0 + k0 + 16 + half * 8);
    ua1.h[0] = *(const v8bf*)(arow1 + k0 + half * 8);
    ua1.h[1] = *(const v8bf*)(arow1 + k0 + 16 + half * 8);
    acc0 = __builtin_amdgcn_wmma_f32_16x16x32_bf16(false, ua0.v, false, ub.v,
                                                   (short)0, acc0, false, false);
    acc1 = __builtin_amdgcn_wmma_f32_16x16x32_bf16(false, ua1.v, false, ub.v,
                                                   (short)0, acc1, false, false);
  }
  // C/D layout: VGPR r -> M = mbase + half*8 + r, col = n0 + l15
  int col = n0 + l15;
  float bv = bias ? bias[col] : 0.f;
  for (int r = 0; r < 8; ++r) {
    float v0 = acc0[r] + bv;
    if (act) v0 = tanhf(v0);
    C[(size_t)(m0 + half * 8 + r) * ldc + col] = v0;
    float v1 = acc1[r] + bv;
    if (act) v1 = tanhf(v1);
    C[(size_t)(m0 + 16 + half * 8 + r) * ldc + col] = v1;
  }
}

// ---------------------------------------------------------------- GRU elementwise
__global__ void k_gru(const float* __restrict__ gi, const float* __restrict__ gh,
                      const float* __restrict__ bih, const float* __restrict__ bhh,
                      const float* __restrict__ h, float* __restrict__ h2,
                      bf16_t* __restrict__ h2_bf) {
  int i = blockIdx.x * blockDim.x + threadIdx.x;
  if (i >= B_ * U_) return;
  int b = i / U_, u = i % U_;
  const float* gib = gi + (size_t)b * 3 * U_;
  const float* ghb = gh + (size_t)b * 3 * U_;
  float ir = gib[u]          + bih[u],          hr = ghb[u]          + bhh[u];
  float iz = gib[U_ + u]     + bih[U_ + u],     hz = ghb[U_ + u]     + bhh[U_ + u];
  float in_= gib[2*U_ + u]   + bih[2*U_ + u],   hn = ghb[2*U_ + u]   + bhh[2*U_ + u];
  float r = 1.f / (1.f + expf(-(ir + hr)));
  float z = 1.f / (1.f + expf(-(iz + hz)));
  float n = tanhf(in_ + r * hn);
  float hv = (1.f - z) * n + z * h[i];
  h2[i]    = hv;
  h2_bf[i] = (bf16_t)hv;
}

// ---------------------------------------------------------------- packers
// rows r = b*S + s, cols [outs_f | outs_b] -> bf16 (B*S x 2U)
__global__ void k_pack_enc(const float* __restrict__ of, const float* __restrict__ ob,
                           bf16_t* __restrict__ out) {
  long i = (long)blockIdx.x * blockDim.x + threadIdx.x;
  long tot = (long)B_ * S_ * 2 * U_;
  if (i >= tot) return;
  int u2 = (int)(i % (2 * U_));
  long r = i / (2 * U_);
  int s = (int)(r % S_), b = (int)(r / S_);
  float v = (u2 < U_) ? of[((size_t)s * B_ + b) * U_ + u2]
                      : ob[((size_t)s * B_ + b) * U_ + (u2 - U_)];
  out[i] = (bf16_t)v;
}

// concat(hf, hb) -> bf16 (B x 2U);  hf = outs_f[S-1], hb = outs_b[0]
__global__ void k_cat_hfhb(const float* __restrict__ of, const float* __restrict__ ob,
                           bf16_t* __restrict__ out) {
  int i = blockIdx.x * blockDim.x + threadIdx.x;
  if (i >= B_ * 2 * U_) return;
  int u2 = i % (2 * U_), b = i / (2 * U_);
  float v = (u2 < U_) ? of[((size_t)(S_ - 1) * B_ + b) * U_ + u2]
                      : ob[(size_t)b * U_ + (u2 - U_)];
  out[i] = (bf16_t)v;
}

// scores[b,s] = sum_a tanh(st_proj[b,a] + enc_proj[b*S+s,a] + attn_b[a])
__global__ void k_scores(const float* __restrict__ stp, const float* __restrict__ ep,
                         const float* __restrict__ ab, float* __restrict__ sc) {
  int s = blockIdx.x, b = blockIdx.y, t = threadIdx.x;
  __shared__ float red[256];
  const float* eprow = ep + ((size_t)b * S_ + s) * A_;
  float acc = 0.f;
  for (int a = t; a < A_; a += 256)
    acc += tanhf(stp[(size_t)b * A_ + a] + eprow[a] + ab[a]);
  red[t] = acc; __syncthreads();
  for (int off = 128; off > 0; off >>= 1) {
    if (t < off) red[t] += red[t + off];
    __syncthreads();
  }
  if (t == 0) sc[(size_t)b * S_ + s] = red[0];
}

__global__ void k_softmax64(const float* __restrict__ sc, float* __restrict__ wts) {
  int b = blockIdx.x, s = threadIdx.x;     // 64 threads
  __shared__ float buf[64];
  float v = sc[(size_t)b * S_ + s];
  buf[s] = v; __syncthreads();
  for (int off = 32; off > 0; off >>= 1) {
    if (s < off) buf[s] = fmaxf(buf[s], buf[s + off]);
    __syncthreads();
  }
  float m = buf[0]; __syncthreads();
  float e = expf(v - m);
  buf[s] = e; __syncthreads();
  for (int off = 32; off > 0; off >>= 1) {
    if (s < off) buf[s] += buf[s + off];
    __syncthreads();
  }
  wts[(size_t)b * S_ + s] = e / buf[0];
}

// ctx[b,u2] = sum_s wts[b,s] * enc_bt[b,s,u2]
__global__ void k_ctx(const float* __restrict__ wts, const float* __restrict__ of,
                      const float* __restrict__ ob, float* __restrict__ ctx) {
  int u2 = blockIdx.x * 256 + threadIdx.x;   // grid.x = 4
  int b  = blockIdx.y;
  if (u2 >= 2 * U_) return;
  float acc = 0.f;
  for (int s = 0; s < S_; ++s) {
    float v = (u2 < U_) ? of[((size_t)s * B_ + b) * U_ + u2]
                        : ob[((size_t)s * B_ + b) * U_ + (u2 - U_)];
    acc += wts[(size_t)b * S_ + s] * v;
  }
  ctx[(size_t)b * 2 * U_ + u2] = acc;
}

// x = [emb_t | ctx] -> bf16 (B x KP_2UE)
__global__ void k_build_x(const float* __restrict__ embT, const int* __restrict__ tgt,
                          int t, const float* __restrict__ ctx, bf16_t* __restrict__ x) {
  int i = blockIdx.x * blockDim.x + threadIdx.x;
  if (i >= B_ * KP_2UE) return;
  int k = i % KP_2UE, b = i / KP_2UE;
  float v = 0.f;
  if (k < E_)              v = embT[(size_t)tgt[t * B_ + b] * E_ + k];
  else if (k < E_ + 2*U_)  v = ctx[(size_t)b * 2 * U_ + (k - E_)];
  x[i] = (bf16_t)v;
}

// z = [st2 | ctx | emb_t] -> bf16 (B x KP_3UE)
__global__ void k_build_z(const float* __restrict__ st2, const float* __restrict__ ctx,
                          const float* __restrict__ embT, const int* __restrict__ tgt,
                          int t, bf16_t* __restrict__ z) {
  int i = blockIdx.x * blockDim.x + threadIdx.x;
  if (i >= B_ * KP_3UE) return;
  int k = i % KP_3UE, b = i / KP_3UE;
  float v = 0.f;
  if (k < U_)            v = st2[(size_t)b * U_ + k];
  else if (k < 3 * U_)   v = ctx[(size_t)b * 2 * U_ + (k - U_)];
  else if (k < 3*U_+E_)  v = embT[(size_t)tgt[t * B_ + b] * E_ + (k - 3 * U_)];
  z[i] = (bf16_t)v;
}

// ---------------------------------------------------------------- launch
static inline dim3 gemm_grid(int M, int N) { return dim3((N + 127) / 128, (M + 31) / 32); }

extern "C" void kernel_launch(void* const* d_in, const int* in_sizes, int n_in,
                              void* d_out, int out_size, void* d_ws, size_t ws_size,
                              hipStream_t stream) {
  const float* emb_src   = (const float*)d_in[0];
  const float* emb_trg   = (const float*)d_in[1];
  const float* eWih_f    = (const float*)d_in[2];
  const float* eWhh_f    = (const float*)d_in[3];
  const float* ebih_f    = (const float*)d_in[4];
  const float* ebhh_f    = (const float*)d_in[5];
  const float* eWih_b    = (const float*)d_in[6];
  const float* eWhh_b    = (const float*)d_in[7];
  const float* ebih_b    = (const float*)d_in[8];
  const float* ebhh_b    = (const float*)d_in[9];
  const float* eWfc      = (const float*)d_in[10];
  const float* ebfc      = (const float*)d_in[11];
  const float* attn_W    = (const float*)d_in[12];
  const float* attn_b    = (const float*)d_in[13];
  const float* dWih      = (const float*)d_in[14];
  const float* dWhh      = (const float*)d_in[15];
  const float* dbih      = (const float*)d_in[16];
  const float* dbhh      = (const float*)d_in[17];
  const float* dWfc      = (const float*)d_in[18];
  const float* dbfc      = (const float*)d_in[19];
  const int*   source    = (const int*)d_in[20];
  const int*   target    = (const int*)d_in[21];
  float*       out       = (float*)d_out;

  // workspace carve-up (256B aligned)
  char* base = (char*)d_ws;
  size_t off = 0;
  auto alloc = [&](size_t bytes) -> char* {
    char* p = base + off;
    off = (off + bytes + 255) & ~(size_t)255;
    return p;
  };
  bf16_t* eWihf_bf = (bf16_t*)alloc((size_t)KP_3U * KP_E   * 2);
  bf16_t* eWhhf_bf = (bf16_t*)alloc((size_t)KP_3U * KP_U   * 2);
  bf16_t* eWihb_bf = (bf16_t*)alloc((size_t)KP_3U * KP_E   * 2);
  bf16_t* eWhhb_bf = (bf16_t*)alloc((size_t)KP_3U * KP_U   * 2);
  bf16_t* eWfc_bf  = (bf16_t*)alloc((size_t)U_    * KP_2U  * 2);
  bf16_t* aWd_bf   = (bf16_t*)alloc((size_t)A_    * KP_U   * 2);
  bf16_t* aWe_bf   = (bf16_t*)alloc((size_t)A_    * KP_2U  * 2);
  bf16_t* dWih_bf  = (bf16_t*)alloc((size_t)KP_3U * KP_2UE * 2);
  bf16_t* dWhh_bf  = (bf16_t*)alloc((size_t)KP_3U * KP_U   * 2);
  bf16_t* dWfc_bf  = (bf16_t*)alloc((size_t)VT_   * KP_3UE * 2);   // 111 MB, fits L2
  bf16_t* srcE_bf  = (bf16_t*)alloc((size_t)S_ * B_ * KP_E * 2);
  float*  gi_f     = (float*) alloc((size_t)S_ * B_ * KP_3U * 4);
  float*  gi_b     = (float*) alloc((size_t)S_ * B_ * KP_3U * 4);
  float*  outs_f   = (float*) alloc((size_t)S_ * B_ * U_ * 4);
  float*  outs_b   = (float*) alloc((size_t)S_ * B_ * U_ * 4);
  float*  h_zero   = (float*) alloc((size_t)B_ * U_ * 4);
  bf16_t* h_bf     = (bf16_t*)alloc((size_t)B_ * U_ * 2);
  float*  gh_buf   = (float*) alloc((size_t)B_ * KP_3U * 4);
  float*  gi_dec   = (float*) alloc((size_t)B_ * KP_3U * 4);
  bf16_t* cat_bf   = (bf16_t*)alloc((size_t)B_ * KP_2U * 2);
  float*  state_f  = (float*) alloc((size_t)B_ * U_ * 4);
  bf16_t* st_bf    = (bf16_t*)alloc((size_t)B_ * U_ * 2);
  bf16_t* encP_bf  = (bf16_t*)alloc((size_t)B_ * S_ * KP_2U * 2);
  float*  enc_proj = (float*) alloc((size_t)B_ * S_ * A_ * 4);
  float*  st_proj  = (float*) alloc((size_t)B_ * A_ * 4);
  float*  scores   = (float*) alloc((size_t)B_ * S_ * 4);
  float*  wts      = (float*) alloc((size_t)B_ * S_ * 4);
  float*  ctx      = (float*) alloc((size_t)B_ * 2 * U_ * 4);
  bf16_t* x_bf     = (bf16_t*)alloc((size_t)B_ * KP_2UE * 2);
  bf16_t* z_bf     = (bf16_t*)alloc((size_t)B_ * KP_3UE * 2);
  (void)ws_size; (void)n_in; (void)in_sizes; (void)out_size;

  auto cvt = [&](const float* in, bf16_t* o, int N, int K, int Kpad, int ld, int co) {
    long tot = (long)N * Kpad;
    k_cvt_bf16<<<dim3((unsigned)((tot + 255) / 256)), 256, 0, stream>>>(in, o, N, K, Kpad, ld, co);
  };

  // ---- weight conversions (fp32 -> padded bf16), once per launch
  cvt(eWih_f, eWihf_bf, KP_3U, E_,   KP_E,   E_,     0);
  cvt(eWhh_f, eWhhf_bf, KP_3U, U_,   KP_U,   U_,     0);
  cvt(eWih_b, eWihb_bf, KP_3U, E_,   KP_E,   E_,     0);
  cvt(eWhh_b, eWhhb_bf, KP_3U, U_,   KP_U,   U_,     0);
  cvt(eWfc,   eWfc_bf,  U_,    2*U_, KP_2U,  2*U_,   0);
  cvt(attn_W, aWd_bf,   A_,    U_,   KP_U,   3*U_,   0);     // Wd = attn_W[:, :U]
  cvt(attn_W, aWe_bf,   A_,    2*U_, KP_2U,  3*U_,   U_);    // We = attn_W[:, U:]
  cvt(dWih,   dWih_bf,  KP_3U, 2*U_+E_, KP_2UE, 2*U_+E_, 0);
  cvt(dWhh,   dWhh_bf,  KP_3U, U_,   KP_U,   U_,     0);
  cvt(dWfc,   dWfc_bf,  VT_,   3*U_+E_, KP_3UE, 3*U_+E_, 0);

  // ---- zero init
  k_fill_f32<<<dim3((B_*U_ + 255) / 256), 256, 0, stream>>>(h_zero, B_*U_, 0.f);
  k_fill_f32<<<dim3((B_*VT_ + 255) / 256), 256, 0, stream>>>(out, (long)B_*VT_, 0.f); // zero_row

  // ---- encoder input GEMMs, batched over all timesteps (time-independent)
  {
    long tot = (long)S_ * B_ * KP_E;
    k_gather_emb<<<dim3((unsigned)((tot + 255) / 256)), 256, 0, stream>>>(
        emb_src, source, srcE_bf, S_*B_, E_, KP_E);
  }
  k_gemm_bf16<<<gemm_grid(S_*B_, KP_3U), 256, 0, stream>>>(
      srcE_bf, KP_E, eWihf_bf, KP_E, gi_f, KP_3U, nullptr, S_*B_, KP_3U, KP_E, 0);
  k_gemm_bf16<<<gemm_grid(S_*B_, KP_3U), 256, 0, stream>>>(
      srcE_bf, KP_E, eWihb_bf, KP_E, gi_b, KP_3U, nullptr, S_*B_, KP_3U, KP_E, 0);

  // ---- encoder forward recurrence
  k_fill_f32<<<dim3((B_*U_/2 + 255) / 256), 256, 0, stream>>>((float*)h_bf, B_*U_/2, 0.f);
  for (int s = 0; s < S_; ++s) {
    k_gemm_bf16<<<gemm_grid(B_, KP_3U), 256, 0, stream>>>(
        h_bf, KP_U, eWhhf_bf, KP_U, gh_buf, KP_3U, nullptr, B_, KP_3U, KP_U, 0);
    const float* hprev = (s == 0) ? h_zero : (outs_f + (size_t)(s - 1) * B_ * U_);
    k_gru<<<dim3((B_*U_ + 255) / 256), 256, 0, stream>>>(
        gi_f + (size_t)s * B_ * KP_3U, gh_buf, ebih_f, ebhh_f,
        hprev, outs_f + (size_t)s * B_ * U_, h_bf);
  }
  // ---- encoder backward recurrence (step t processes original position S-1-t)
  k_fill_f32<<<dim3((B_*U_/2 + 255) / 256), 256, 0, stream>>>((float*)h_bf, B_*U_/2, 0.f);
  for (int t = 0; t < S_; ++t) {
    int s = S_ - 1 - t;
    k_gemm_bf16<<<gemm_grid(B_, KP_3U), 256, 0, stream>>>(
        h_bf, KP_U, eWhhb_bf, KP_U, gh_buf, KP_3U, nullptr, B_, KP_3U, KP_U, 0);
    const float* hprev = (t == 0) ? h_zero : (outs_b + (size_t)(s + 1) * B_ * U_);
    k_gru<<<dim3((B_*U_ + 255) / 256), 256, 0, stream>>>(
        gi_b + (size_t)s * B_ * KP_3U, gh_buf, ebih_b, ebhh_b,
        hprev, outs_b + (size_t)s * B_ * U_, h_bf);
  }

  // ---- initial decoder state: tanh([hf|hb] @ Wfc^T + bfc)
  k_cat_hfhb<<<dim3((B_*2*U_ + 255) / 256), 256, 0, stream>>>(outs_f, outs_b, cat_bf);
  k_gemm_bf16<<<gemm_grid(B_, U_), 256, 0, stream>>>(
      cat_bf, KP_2U, eWfc_bf, KP_2U, state_f, U_, ebfc, B_, U_, KP_2U, 1 /*tanh*/);
  cvt(state_f, st_bf, B_, U_, KP_U, U_, 0);

  // ---- attention precompute: enc_proj = enc_bt @ We^T  (decoder-step invariant)
  {
    long tot = (long)B_ * S_ * KP_2U;
    k_pack_enc<<<dim3((unsigned)((tot + 255) / 256)), 256, 0, stream>>>(outs_f, outs_b, encP_bf);
  }
  k_gemm_bf16<<<gemm_grid(B_*S_, A_), 256, 0, stream>>>(
      encP_bf, KP_2U, aWe_bf, KP_2U, enc_proj, A_, nullptr, B_*S_, A_, KP_2U, 0);

  // ---- decoder loop (31 steps)
  for (int t = 0; t < T_ - 1; ++t) {
    k_gemm_bf16<<<gemm_grid(B_, A_), 256, 0, stream>>>(
        st_bf, KP_U, aWd_bf, KP_U, st_proj, A_, nullptr, B_, A_, KP_U, 0);
    k_scores<<<dim3(S_, B_), 256, 0, stream>>>(st_proj, enc_proj, attn_b, scores);
    k_softmax64<<<dim3(B_), 64, 0, stream>>>(scores, wts);
    k_ctx<<<dim3(4, B_), 256, 0, stream>>>(wts, outs_f, outs_b, ctx);
    k_build_x<<<dim3((B_*KP_2UE + 255) / 256), 256, 0, stream>>>(emb_trg, target, t, ctx, x_bf);
    k_gemm_bf16<<<gemm_grid(B_, KP_3U), 256, 0, stream>>>(
        x_bf, KP_2UE, dWih_bf, KP_2UE, gi_dec, KP_3U, nullptr, B_, KP_3U, KP_2UE, 0);
    k_gemm_bf16<<<gemm_grid(B_, KP_3U), 256, 0, stream>>>(
        st_bf, KP_U, dWhh_bf, KP_U, gh_buf, KP_3U, nullptr, B_, KP_3U, KP_U, 0);
    k_gru<<<dim3((B_*U_ + 255) / 256), 256, 0, stream>>>(
        gi_dec, gh_buf, dbih, dbhh, state_f, state_f, st_bf);  // in-place state update
    k_build_z<<<dim3((B_*KP_3UE + 255) / 256), 256, 0, stream>>>(
        state_f, ctx, emb_trg, target, t, z_bf);
    // dominant GEMM: 32 x 30000 x 1856, bf16 weights resident in 192MB L2,
    // each weight fragment now feeds both M-tiles (single pass over dWfc per step)
    k_gemm_bf16<<<gemm_grid(B_, VT_), 256, 0, stream>>>(
        z_bf, KP_3UE, dWfc_bf, KP_3UE, out + (size_t)(t + 1) * B_ * VT_, VT_,
        dbfc, B_, VT_, KP_3UE, 0);
  }
}
